// Swin3DBlock_NAS_78735340470546
// MI455X (gfx1250) — compile-verified, matrix operand order checked by reading, not run on hardware
//
#include <hip/hip_runtime.h>

// ---------------------------------------------------------------------------
// Swin3D NAS block for gfx1250 (MI455X). GEMM-shaped math on
// v_wmma_f32_16x16x32_bf16; fragment I/O via 16-byte vector loads;
// V staged into LDS with global_load_async_to_lds_b128 (ASYNCcnt);
// attention bias/mask folded into precomputed streaming tables.
// ---------------------------------------------------------------------------

typedef __attribute__((ext_vector_type(16))) __bf16 v16bf;
typedef __attribute__((ext_vector_type(8)))  float  v8f;
typedef __attribute__((ext_vector_type(8)))  unsigned short us8;

#define BATCH 2
#define CDIM 96
#define DD 8
#define HH 56
#define WW 56
#define NPOS (DD * HH * WW)            // 25088
#define NTOK 392                        // 8*7*7 tokens per window
#define NPAD 400                        // padded to 25 row-tiles of 16
#define NWIN 128                        // 2 * 64 windows
#define TBL (NPAD * NPAD)               // 160000 entries per table slot

__device__ __forceinline__ unsigned short f2bf(float f) {
  union { __bf16 b; unsigned short u; } a;
  a.b = (__bf16)f;  // native v_cvt f32->bf16 (RNE)
  return a.u;
}
__device__ __forceinline__ float bf2f(unsigned short h) {
  union { unsigned u; float f; } a; a.u = ((unsigned)h) << 16;
  return a.f;
}

__device__ __forceinline__ v8f wmma_bf16(v16bf a, v16bf b, v8f c) {
  return __builtin_amdgcn_wmma_f32_16x16x32_bf16(false, a, false, b, (short)0, c,
                                                 false, false);
}

union frag_u {
  v16bf v;
  us8 h[2];
};

// Full 16x32 fragment: per-lane data = two contiguous 8-elem bf16 runs at
// kbase and kbase+16, kbase = k0 + (lane>=16 ? 8 : 0)  -> two b128 loads.
__device__ __forceinline__ v16bf load_frag_full(const unsigned short* base,
                                                int ld, int r0, int k0) {
  int lane = threadIdx.x & 31;
  const unsigned short* p = base + (size_t)(r0 + (lane & 15)) * (size_t)ld +
                            k0 + ((lane >> 4) ? 8 : 0);
  frag_u f;
  f.h[0] = *(const us8*)(p);
  f.h[1] = *(const us8*)(p + 16);
  return f.v;
}

// Half fragment: only kk in [k0, k0+16) valid (dh=16 heads, PV tail).
__device__ __forceinline__ v16bf load_frag_half(const unsigned short* base,
                                                int ld, int r0, int k0) {
  int lane = threadIdx.x & 31;
  const unsigned short* p = base + (size_t)(r0 + (lane & 15)) * (size_t)ld +
                            k0 + ((lane >> 4) ? 8 : 0);
  frag_u f;
  us8 z = {0, 0, 0, 0, 0, 0, 0, 0};
  f.h[0] = *(const us8*)(p);
  f.h[1] = z;
  return f.v;
}

// A-fragment for conv rows = 16 consecutive W positions of x2t (B,D,H,W,C)
__device__ __forceinline__ v16bf load_a_conv(const unsigned short* x2t, int b,
                                             int dp, int hp, int w0, int k0) {
  int lane = threadIdx.x & 31;
  int w = w0 + (lane & 15);
  frag_u f;
  if ((unsigned)w < (unsigned)WW) {
    const unsigned short* p =
        x2t + ((size_t)(((b * DD + dp) * HH + hp) * WW + w)) * CDIM + k0 +
        ((lane >> 4) ? 8 : 0);
    f.h[0] = *(const us8*)(p);
    f.h[1] = *(const us8*)(p + 16);
  } else {
    us8 z = {0, 0, 0, 0, 0, 0, 0, 0};
    f.h[0] = z;
    f.h[1] = z;
  }
  return f.v;
}

// ---------------------------------------------------------------------------
__global__ void k_convert_weights(const float* qkvw0, const float* qkvw1,
                                  const float* projw0, const float* projw1,
                                  const float* conv3w, const float* conv1w,
                                  unsigned short* o_qkv0, unsigned short* o_qkv1,
                                  unsigned short* o_p0, unsigned short* o_p1,
                                  unsigned short* o_w3t, unsigned short* o_w1t) {
  int i = blockIdx.x * blockDim.x + threadIdx.x;
  int stride = gridDim.x * blockDim.x;
  for (int t = i; t < 3 * CDIM * CDIM; t += stride) {
    o_qkv0[t] = f2bf(qkvw0[t]);
    o_qkv1[t] = f2bf(qkvw1[t]);
  }
  for (int t = i; t < CDIM * CDIM; t += stride) {
    o_p0[t] = f2bf(projw0[t]);
    o_p1[t] = f2bf(projw1[t]);
    o_w1t[t] = f2bf(conv1w[t]);
  }
  for (int t = i; t < 27 * CDIM * CDIM; t += stride) {
    int tap = t / (CDIM * CDIM);
    int r = t % (CDIM * CDIM);
    int co = r / CDIM, ci = r % CDIM;
    o_w3t[t] = f2bf(conv3w[(co * CDIM + ci) * 27 + tap]);
  }
}

__global__ void k_zero_pad(unsigned short* xw) {
  int i = blockIdx.x * blockDim.x + threadIdx.x;
  if (i < NWIN * 8 * CDIM) {
    int w = i / (8 * CDIM);
    int r = i % (8 * CDIM);
    int n = NTOK + r / CDIM;
    int c = r % CDIM;
    xw[((size_t)w * NPAD + n) * CDIM + c] = 0;
  }
}

// ---------------------------------------------------------------------------
// Precomputed additive score tables.
// biastab[slot][qi][j]: relative-position bias (slot 0..2 = cfg0 heads,
// 3..8 = cfg1 heads); j >= 392 -> -1e30 (key padding).
// masktab[variant][qi][j]: shift-window mask; variant = (hI==7)*2 + (wI==7).
// ---------------------------------------------------------------------------
__global__ void k_bias_tab(const float* rpb0, const float* rpb1,
                           float* biastab) {
  int idx = blockIdx.x * blockDim.x + threadIdx.x;
  if (idx >= 9 * TBL) return;
  int slot = idx / TBL;
  int r = idx % TBL;
  int qi = r / NPAD, j = r % NPAD;
  int cfg = (slot < 3) ? 0 : 1;
  int h = (cfg == 0) ? slot : slot - 3;
  int hcnt = (cfg == 0) ? 3 : 6;
  const float* rpb = (cfg == 0) ? rpb0 : rpb1;
  float v;
  if (j >= NTOK) {
    v = -1e30f;
  } else {
    int qc = (qi < NTOK) ? qi : 0;
    int di = qc / 49, hi = (qc % 49) / 7, wi = qc % 7;
    int dj = j / 49, hj = (j % 49) / 7, wj = j % 7;
    int ridx = (di - dj + 7) * 169 + (hi - hj + 6) * 13 + (wi - wj + 6);
    v = rpb[ridx * hcnt + h];
  }
  biastab[idx] = v;
}

__global__ void k_mask_tab(float* masktab) {
  int idx = blockIdx.x * blockDim.x + threadIdx.x;
  if (idx >= 4 * TBL) return;
  int var = idx / TBL;
  int r = idx % TBL;
  int qi = r / NPAD, j = r % NPAD;
  int hI = (var & 2) ? 7 : 0, wI = (var & 1) ? 7 : 0;
  float v = 0.f;
  if (j < NTOK && qi < NTOK) {
    int di = qi / 49, hi = (qi % 49) / 7, wi = qi % 7;
    int dj = j / 49, hj = (j % 49) / 7, wj = j % 7;
    int ghi = hI * 7 + hi, gwi = wI * 7 + wi;
    int ghj = hI * 7 + hj, gwj = wI * 7 + wj;
    int li = (di < 4 ? 0 : 1) * 9 +
             (ghi < 49 ? 0 : (ghi < 53 ? 1 : 2)) * 3 +
             (gwi < 49 ? 0 : (gwi < 53 ? 1 : 2));
    int lj = (dj < 4 ? 0 : 1) * 9 +
             (ghj < 49 ? 0 : (ghj < 53 ? 1 : 2)) * 3 +
             (gwj < 49 ? 0 : (gwj < 53 ? 1 : 2));
    v = (li != lj) ? -100.f : 0.f;
  }
  masktab[idx] = v;
}

// ---------------------------------------------------------------------------
__global__ void k_ln1_partition(const float* x, const float* ln1w,
                                const float* ln1b, unsigned short* xw) {
  int p = blockIdx.x * blockDim.x + threadIdx.x;
  if (p >= BATCH * NPOS) return;
  int b = p / NPOS, r = p % NPOS;
  int d2 = r / (HH * WW);
  int r2 = r % (HH * WW);
  int h2 = r2 / WW, w2 = r2 % WW;
  int d = (d2 + 4) & 7, h = (h2 + 3) % HH, w = (w2 + 3) % WW;
  const float* src = x + (size_t)b * CDIM * NPOS + d * (HH * WW) + h * WW + w;
  float mu = 0.f;
  for (int c = 0; c < CDIM; ++c) mu += src[(size_t)c * NPOS];
  mu *= (1.f / CDIM);
  float var = 0.f;
  for (int c = 0; c < CDIM; ++c) {
    float v = src[(size_t)c * NPOS] - mu;
    var += v * v;
  }
  var *= (1.f / CDIM);
  float rs = rsqrtf(var + 1e-5f);
  int widx = b * 64 + (h2 / 7) * 8 + (w2 / 7);
  int n = d2 * 49 + (h2 % 7) * 7 + (w2 % 7);
  unsigned short* dst = xw + ((size_t)widx * NPAD + n) * CDIM;
  for (int c = 0; c < CDIM; ++c) {
    float v = (src[(size_t)c * NPOS] - mu) * rs * ln1w[c] + ln1b[c];
    dst[c] = f2bf(v);
  }
}

// ---------------------------------------------------------------------------
// QKV GEMM: 8 tiles per 256-thread block (one 16x16 tile per wave), K=96.
// ---------------------------------------------------------------------------
__global__ __launch_bounds__(256) void k_gemm_bias(const unsigned short* A,
                                                   int lda,
                                                   const unsigned short* Bc,
                                                   int ldb, const float* bias,
                                                   unsigned short* out, int ldo,
                                                   int ntcols, int ntiles) {
  int wave = threadIdx.x >> 5;
  int id = blockIdx.x * 8 + wave;
  if (id >= ntiles) return;
  int mt = id / ntcols, nt = id % ntcols;
  v8f c = {};
  for (int k0 = 0; k0 < CDIM; k0 += 32) {
    __builtin_prefetch(A + (size_t)(mt * 16 + 16) * lda + k0, 0, 1);
    v16bf a = load_frag_full(A, lda, mt * 16, k0);
    v16bf b = load_frag_full(Bc, ldb, nt * 16, k0);
    c = wmma_bf16(a, b, c);
  }
  int lane = threadIdx.x & 31;
  int n = nt * 16 + (lane & 15);
  int mbase = mt * 16 + ((lane >> 4) ? 8 : 0);
  float bv = bias[n];
#pragma unroll
  for (int r = 0; r < 8; ++r)
    out[(size_t)(mbase + r) * ldo + n] = f2bf(c[r] + bv);
}

// proj GEMM + softmax(alpha_attn) mixture into f32 accumulator
__global__ __launch_bounds__(256) void k_gemm_proj(const unsigned short* A,
                                                   const unsigned short* Bc,
                                                   const float* bias,
                                                   const float* alpha,
                                                   int which, int init,
                                                   float* out, int ntiles) {
  int wave = threadIdx.x >> 5;
  int id = blockIdx.x * 8 + wave;
  if (id >= ntiles) return;
  int mt = id / (CDIM / 16), nt = id % (CDIM / 16);
  v8f c = {};
  for (int k0 = 0; k0 < CDIM; k0 += 32) {
    v16bf a = load_frag_full(A, CDIM, mt * 16, k0);
    v16bf b = load_frag_full(Bc, CDIM, nt * 16, k0);
    c = wmma_bf16(a, b, c);
  }
  float a0 = alpha[0], a1 = alpha[1];
  float m = fmaxf(a0, a1);
  float e0 = __expf(a0 - m), e1 = __expf(a1 - m);
  float aw = ((which == 0) ? e0 : e1) / (e0 + e1);
  int lane = threadIdx.x & 31;
  int n = nt * 16 + (lane & 15);
  int mbase = mt * 16 + ((lane >> 4) ? 8 : 0);
  float bv = bias[n];
#pragma unroll
  for (int r = 0; r < 8; ++r) {
    size_t idx = (size_t)(mbase + r) * CDIM + n;
    float v = aw * (c[r] + bv);
    if (init) out[idx] = v;
    else out[idx] += v;
  }
}

// ---------------------------------------------------------------------------
// Fused per-(window, head) attention.
// ---------------------------------------------------------------------------
union SUnion {
  float S[16 * NPAD];                 // 25600 B score tile
  unsigned short Vstage[NPAD * 32];   // 25600 B async V staging (aliased)
};

__global__ __launch_bounds__(256) void k_attn(const unsigned short* qkv0,
                                              const unsigned short* qkv1,
                                              const float* biastab,
                                              const float* masktab,
                                              unsigned short* ho0,
                                              unsigned short* ho1) {
  __shared__ __align__(16) SUnion uS;
  __shared__ __align__(16) unsigned short P[16 * NPAD];   // 12800 B
  __shared__ __align__(16) unsigned short Vt[32 * NPAD];  // 25600 B  V^T
  __shared__ float rowinv[16];

  int widx = blockIdx.x;
  int y = blockIdx.y;
  int cfg = (y < 3) ? 0 : 1;
  int h = (cfg == 0) ? y : (y - 3);
  int dh = (cfg == 0) ? 32 : 16;
  const unsigned short* qkv = (cfg == 0) ? qkv0 : qkv1;
  unsigned short* ho = (cfg == 0) ? ho0 : ho1;
  const unsigned short* base = qkv + (size_t)widx * NPAD * (3 * CDIM);
  float scale = rsqrtf((float)dh);
  int tid = threadIdx.x;
  int wave = tid >> 5;
  int lane = tid & 31;

  int wb = widx & 63;
  int hI = wb >> 3, wI = wb & 7;
  const float* btab = biastab + (size_t)y * TBL;
  const float* mtab = masktab + (size_t)(((hI == 7) ? 2 : 0) | ((wI == 7) ? 1 : 0)) * TBL;

  // ---- stage V rows (contiguous 2*dh bytes each) into LDS via async b128,
  //      then transpose LDS->LDS into Vt[c][tok]. ----
  {
    int cpr = (dh * 2) / 16;  // 16B chunks per token row: 4 (dh=32) / 2 (dh=16)
    int total = NPAD * cpr;
    for (int i = tid; i < total; i += 256) {
      int tok = i / cpr, sub = i % cpr;
      const unsigned short* g =
          base + (size_t)tok * (3 * CDIM) + 2 * CDIM + h * dh + sub * 8;
      unsigned lds_off = (unsigned)(size_t)(const void*)&uS.Vstage[tok * dh + sub * 8];
      asm volatile("global_load_async_to_lds_b128 %0, %1, off"
                   :: "v"(lds_off), "v"((unsigned long long)(size_t)g)
                   : "memory");
    }
    asm volatile("s_wait_asynccnt 0" ::: "memory");
    __syncthreads();
    for (int i = tid; i < dh * NPAD; i += 256) {
      int c = i / NPAD, tok = i % NPAD;
      Vt[c * NPAD + tok] = uS.Vstage[tok * dh + c];
    }
    __syncthreads();
  }

  for (int qt = 0; qt < 25; ++qt) {
    // ---- S = Q K^T : waves split the 25 key tiles ----
    {
      v16bf aq = (dh == 32) ? load_frag_full(base + h * dh, 3 * CDIM, qt * 16, 0)
                            : load_frag_half(base + h * dh, 3 * CDIM, qt * 16, 0);
      for (int kt = wave; kt < 25; kt += 8) {
        v16bf bb = (dh == 32)
                       ? load_frag_full(base + CDIM + h * dh, 3 * CDIM, kt * 16, 0)
                       : load_frag_half(base + CDIM + h * dh, 3 * CDIM, kt * 16, 0);
        v8f c = {};
        c = wmma_bf16(aq, bb, c);
        int n = kt * 16 + (lane & 15);
        int mb = (lane >> 4) ? 8 : 0;
#pragma unroll
        for (int r = 0; r < 8; ++r) uS.S[(mb + r) * NPAD + n] = c[r];
      }
    }
    __syncthreads();

    // ---- scale + bias + mask from precomputed streaming tables ----
    {
      const float* br = btab + (size_t)qt * 16 * NPAD;
      const float* mr = mtab + (size_t)qt * 16 * NPAD;
      for (int t = tid; t < 16 * NPAD; t += 256)
        uS.S[t] = uS.S[t] * scale + br[t] + mr[t];
    }
    __syncthreads();

    // ---- softmax: 16 lanes per row ----
    {
      int i = tid >> 4, sub = tid & 15;
      float mx = -1e30f;
      for (int j = sub; j < NPAD; j += 16) mx = fmaxf(mx, uS.S[i * NPAD + j]);
#pragma unroll
      for (int o = 8; o >= 1; o >>= 1) mx = fmaxf(mx, __shfl_xor(mx, o, 16));
      float sum = 0.f;
      for (int j = sub; j < NPAD; j += 16) {
        float e = __expf(uS.S[i * NPAD + j] - mx);
        P[i * NPAD + j] = f2bf(e);
        sum += e;
      }
#pragma unroll
      for (int o = 8; o >= 1; o >>= 1) sum += __shfl_xor(sum, o, 16);
      if (sub == 0) rowinv[i] = 1.f / sum;
    }
    __syncthreads();

    // ---- out = P V : one wave per 16-col tile of dh; 12 full + 1 half ----
    int ntiles = dh >> 4;
    if (wave < ntiles) {
      int nt = wave;
      v8f c = {};
      for (int ks = 0; ks < 12; ++ks) {
        v16bf a = load_frag_full(P, NPAD, 0, ks * 32);
        v16bf bb = load_frag_full(Vt, NPAD, nt * 16, ks * 32);
        c = wmma_bf16(a, bb, c);
      }
      {
        v16bf a = load_frag_half(P, NPAD, 0, 384);
        v16bf bb = load_frag_half(Vt, NPAD, nt * 16, 384);
        c = wmma_bf16(a, bb, c);
      }
      int n = nt * 16 + (lane & 15);
      int mb = (lane >> 4) ? 8 : 0;
#pragma unroll
      for (int r = 0; r < 8; ++r) {
        int row = mb + r;
        int tok = qt * 16 + row;
        if (tok < NTOK) {
          float v = c[r] * rowinv[row];
          ho[((size_t)widx * NPAD + tok) * CDIM + h * dh + n] = f2bf(v);
        }
      }
    }
    __syncthreads();
  }
}

// ---------------------------------------------------------------------------
__global__ void k_reverse_ln2(const float* x, const float* comb,
                              const float* ln2w, const float* ln2b, float* x1,
                              unsigned short* x2t) {
  int p = blockIdx.x * blockDim.x + threadIdx.x;
  if (p >= BATCH * NPOS) return;
  int b = p / NPOS, r = p % NPOS;
  int d = r / (HH * WW);
  int r2 = r % (HH * WW);
  int h = r2 / WW, w = r2 % WW;
  int d2 = (d + 4) & 7;
  int h2 = (h + HH - 3) % HH;
  int w2 = (w + WW - 3) % WW;
  int widx = b * 64 + (h2 / 7) * 8 + (w2 / 7);
  int n = d2 * 49 + (h2 % 7) * 7 + (w2 % 7);
  const float* av = comb + ((size_t)widx * NPAD + n) * CDIM;
  const float* xs = x + (size_t)b * CDIM * NPOS + d * (HH * WW) + h * WW + w;
  float* x1s = x1 + (size_t)b * CDIM * NPOS + d * (HH * WW) + h * WW + w;
  float vals[CDIM];
  float mu = 0.f;
  for (int c = 0; c < CDIM; ++c) {
    float v = xs[(size_t)c * NPOS] + av[c];
    vals[c] = v;
    mu += v;
  }
  mu *= (1.f / CDIM);
  float var = 0.f;
  for (int c = 0; c < CDIM; ++c) {
    float t = vals[c] - mu;
    var += t * t;
  }
  var *= (1.f / CDIM);
  float rs = rsqrtf(var + 1e-5f);
  unsigned short* x2s = x2t + (size_t)p * CDIM;
  for (int c = 0; c < CDIM; ++c) {
    x1s[(size_t)c * NPOS] = vals[c];
    x2s[c] = f2bf((vals[c] - mu) * rs * ln2w[c] + ln2b[c]);
  }
}

// ---------------------------------------------------------------------------
__global__ __launch_bounds__(256) void k_ffn(const unsigned short* x2t,
                                             const unsigned short* w3t,
                                             const unsigned short* w1t,
                                             const float* conv3b,
                                             const float* conv1b,
                                             const float* dww, const float* dwb,
                                             const float* alphaf,
                                             const float* x1, float* out) {
  __shared__ float O3[16 * CDIM];
  __shared__ float O1[16 * CDIM];
  int bid = blockIdx.x;
  int wt = bid & 3;
  int rest = bid >> 2;
  int h0 = rest % HH;
  int rest2 = rest / HH;
  int d = rest2 & 7;
  int b = rest2 >> 3;
  int tid = threadIdx.x;
  int wave = tid >> 5;
  int lane = tid & 31;

  if (wave < 6) {
    int nt = wave;
    v8f acc3 = {};
    v8f acc1 = {};
    for (int tap = 0; tap < 27; ++tap) {
      int dd = tap / 9, hh = (tap / 3) % 3, ww = tap % 3;
      int dp = d + dd - 1, hp = h0 + hh - 1;
      if (dp < 0 || dp >= DD || hp < 0 || hp >= HH) continue;
      for (int ks = 0; ks < 3; ++ks) {
        v16bf a = load_a_conv(x2t, b, dp, hp, wt * 16 + ww - 1, ks * 32);
        v16bf bb = load_frag_full(w3t + (size_t)tap * CDIM * CDIM, CDIM,
                                  nt * 16, ks * 32);
        acc3 = wmma_bf16(a, bb, acc3);
      }
    }
    for (int ks = 0; ks < 3; ++ks) {
      v16bf a = load_a_conv(x2t, b, d, h0, wt * 16, ks * 32);
      v16bf bb = load_frag_full(w1t, CDIM, nt * 16, ks * 32);
      acc1 = wmma_bf16(a, bb, acc1);
    }
    int n = nt * 16 + (lane & 15);
    int mb = (lane >> 4) ? 8 : 0;
#pragma unroll
    for (int r = 0; r < 8; ++r) {
      O3[(mb + r) * CDIM + n] = acc3[r];
      O1[(mb + r) * CDIM + n] = acc1[r];
    }
  }
  __syncthreads();

  float a0 = alphaf[0], a1 = alphaf[1], a2 = alphaf[2], a3 = alphaf[3];
  float mx = fmaxf(fmaxf(a0, a1), fmaxf(a2, a3));
  float e0 = __expf(a0 - mx), e1 = __expf(a1 - mx);
  float e2 = __expf(a2 - mx), e3 = __expf(a3 - mx);
  float inv = 1.f / (e0 + e1 + e2 + e3);
  float f0 = e0 * inv, f1 = e1 * inv, f2 = e2 * inv, f3 = e3 * inv;

  for (int t = tid; t < 16 * CDIM; t += 256) {
    int m = t / CDIM, c = t % CDIM;
    int w = wt * 16 + m;
    if (w >= WW) continue;
    float dwv = 0.f;
    for (int tap = 0; tap < 27; ++tap) {
      int dd = tap / 9, hh = (tap / 3) % 3, ww = tap % 3;
      int dp = d + dd - 1, hp = h0 + hh - 1, wp = w + ww - 1;
      if (dp < 0 || dp >= DD || hp < 0 || hp >= HH || wp < 0 || wp >= WW)
        continue;
      dwv += bf2f(x2t[((size_t)(((b * DD + dp) * HH + hp) * WW + wp)) * CDIM + c]) *
             dww[c * 27 + tap];
    }
    float x2v = bf2f(x2t[((size_t)(((b * DD + d) * HH + h0) * WW + w)) * CDIM + c]);
    float ffn = f0 * (O3[t] + conv3b[c]) + f1 * (O1[t] + conv1b[c]) +
                f2 * (dwv + dwb[c]) + f3 * x2v;
    size_t oi = ((size_t)(b * CDIM + c) * DD + d) * (HH * WW) + h0 * WW + w;
    out[oi] = x1[oi] + ffn;
  }
}

// ---------------------------------------------------------------------------
extern "C" void kernel_launch(void* const* d_in, const int* in_sizes, int n_in,
                              void* d_out, int out_size, void* d_ws,
                              size_t ws_size, hipStream_t stream) {
  const float* x        = (const float*)d_in[0];
  const float* ln1_w    = (const float*)d_in[1];
  const float* ln1_b    = (const float*)d_in[2];
  const float* ln2_w    = (const float*)d_in[3];
  const float* ln2_b    = (const float*)d_in[4];
  const float* qkv_w0   = (const float*)d_in[5];
  const float* qkv_b0   = (const float*)d_in[6];
  const float* proj_w0  = (const float*)d_in[7];
  const float* proj_b0  = (const float*)d_in[8];
  const float* rpb0     = (const float*)d_in[9];
  const float* qkv_w1   = (const float*)d_in[10];
  const float* qkv_b1   = (const float*)d_in[11];
  const float* proj_w1  = (const float*)d_in[12];
  const float* proj_b1  = (const float*)d_in[13];
  const float* rpb1     = (const float*)d_in[14];
  const float* alpha_at = (const float*)d_in[15];
  const float* conv3_w  = (const float*)d_in[16];
  const float* conv3_b  = (const float*)d_in[17];
  const float* conv1_w  = (const float*)d_in[18];
  const float* conv1_b  = (const float*)d_in[19];
  const float* dw_w     = (const float*)d_in[20];
  const float* dw_b     = (const float*)d_in[21];
  const float* alpha_ff = (const float*)d_in[22];
  float* out = (float*)d_out;

  size_t off = 0;
  auto carve = [&](size_t bytes) {
    void* p = (char*)d_ws + off;
    off += (bytes + 255) & ~(size_t)255;
    return p;
  };
  unsigned short* w_qkv0 = (unsigned short*)carve(3 * CDIM * CDIM * 2);
  unsigned short* w_qkv1 = (unsigned short*)carve(3 * CDIM * CDIM * 2);
  unsigned short* w_p0   = (unsigned short*)carve(CDIM * CDIM * 2);
  unsigned short* w_p1   = (unsigned short*)carve(CDIM * CDIM * 2);
  unsigned short* w_w1t  = (unsigned short*)carve(CDIM * CDIM * 2);
  unsigned short* w_w3t  = (unsigned short*)carve(27 * CDIM * CDIM * 2);
  unsigned short* xw     = (unsigned short*)carve((size_t)NWIN * NPAD * CDIM * 2);
  unsigned short* qkv0   = (unsigned short*)carve((size_t)NWIN * NPAD * 3 * CDIM * 2);
  unsigned short* qkv1   = (unsigned short*)carve((size_t)NWIN * NPAD * 3 * CDIM * 2);
  unsigned short* ho0    = (unsigned short*)carve((size_t)NWIN * NPAD * CDIM * 2);
  unsigned short* ho1    = (unsigned short*)carve((size_t)NWIN * NPAD * CDIM * 2);
  float* comb            = (float*)carve((size_t)NWIN * NPAD * CDIM * 4);
  float* x1              = (float*)carve((size_t)BATCH * CDIM * NPOS * 4);
  unsigned short* x2t    = (unsigned short*)carve((size_t)BATCH * NPOS * CDIM * 2);
  float* biastab         = (float*)carve((size_t)9 * TBL * 4);
  float* masktab         = (float*)carve((size_t)4 * TBL * 4);

  k_convert_weights<<<256, 256, 0, stream>>>(qkv_w0, qkv_w1, proj_w0, proj_w1,
                                             conv3_w, conv1_w, w_qkv0, w_qkv1,
                                             w_p0, w_p1, w_w3t, w_w1t);
  k_bias_tab<<<(9 * TBL + 255) / 256, 256, 0, stream>>>(rpb0, rpb1, biastab);
  k_mask_tab<<<(4 * TBL + 255) / 256, 256, 0, stream>>>(masktab);
  k_zero_pad<<<(NWIN * 8 * CDIM + 255) / 256, 256, 0, stream>>>(xw);
  k_ln1_partition<<<(BATCH * NPOS + 255) / 256, 256, 0, stream>>>(x, ln1_w,
                                                                  ln1_b, xw);
  // QKV GEMMs: (51200 x 96) @ (96 x 288), 57600 tiles, 8 per block
  {
    int ntiles = (NWIN * NPAD / 16) * (3 * CDIM / 16);
    int blocks = (ntiles + 7) / 8;
    k_gemm_bias<<<blocks, 256, 0, stream>>>(xw, CDIM, w_qkv0, CDIM, qkv_b0,
                                            qkv0, 3 * CDIM, 3 * CDIM / 16,
                                            ntiles);
    k_gemm_bias<<<blocks, 256, 0, stream>>>(xw, CDIM, w_qkv1, CDIM, qkv_b1,
                                            qkv1, 3 * CDIM, 3 * CDIM / 16,
                                            ntiles);
  }
  // fused windowed attention: 128 windows x (3 + 6) heads
  k_attn<<<dim3(NWIN, 9), 256, 0, stream>>>(qkv0, qkv1, biastab, masktab, ho0,
                                            ho1);
  // proj + alpha_attn mixture
  {
    int ntiles = (NWIN * NPAD / 16) * (CDIM / 16);
    int blocks = (ntiles + 7) / 8;
    k_gemm_proj<<<blocks, 256, 0, stream>>>(ho0, w_p0, proj_b0, alpha_at, 0, 1,
                                            comb, ntiles);
    k_gemm_proj<<<blocks, 256, 0, stream>>>(ho1, w_p1, proj_b1, alpha_at, 1, 0,
                                            comb, ntiles);
  }
  k_reverse_ln2<<<(BATCH * NPOS + 255) / 256, 256, 0, stream>>>(x, comb, ln2_w,
                                                                ln2_b, x1, x2t);
  k_ffn<<<BATCH * DD * HH * 4, 256, 0, stream>>>(x2t, w_w3t, w_w1t, conv3_b,
                                                 conv1_b, dw_w, dw_b, alpha_ff,
                                                 x1, out);
}
